// AdvancedTextEncoder_59863254172481
// MI455X (gfx1250) — compile-verified, matrix operand order checked by reading, not run on hardware
//
#include <hip/hip_runtime.h>
#include <hip/hip_bf16.h>
#include <math.h>

// ---------------- model dims ----------------
#define BB 16
#define TT 512
#define EE 1024
#define HH 16
#define LL 8
#define HD 64
#define FF 4096
#define OO 1024
#define BT (BB * TT)   // 8192

typedef __bf16 bf16_t;
typedef __attribute__((ext_vector_type(16))) __bf16 v16bf;
typedef __attribute__((ext_vector_type(4)))  __bf16 bf16x4;
typedef __attribute__((ext_vector_type(8)))  float  v8f;
typedef __attribute__((ext_vector_type(4)))  float  f32x4;
typedef __attribute__((ext_vector_type(4)))  unsigned int u32x4;
typedef __attribute__((ext_vector_type(8)))  int    i32x8;
typedef __attribute__((ext_vector_type(4)))  int    i32x4;

#if defined(__has_builtin)
#if __has_builtin(__builtin_amdgcn_tensor_load_to_lds) && \
    __has_builtin(__builtin_amdgcn_s_wait_tensorcnt)
#define USE_TDM 1
#endif
#endif

// -------------------------------------------------------------------------
// WMMA 16-bit fragment loader (AMD table-order K layout), fully vectorized.
// Lane = 16*g + r (r = M row for A-frag / N row for B-frag).
// Fragment element f[2v+h] holds k = (v<4) ? 8g+2v+h : 16+8g+2(v-4)+h, i.e.
// two contiguous 8-element (16-byte) chunks at k = 8g and k = 16+8g.
// Requires rstride % 8 == 0 and 16-byte aligned base.
// -------------------------------------------------------------------------
union FragU { u32x4 u[2]; v16bf f; };

__device__ __forceinline__ v16bf load_frag_v(const bf16_t* __restrict__ base,
                                             int rstride) {
  const int lane = threadIdx.x & 31;
  const int r = lane & 15, g = lane >> 4;
  const bf16_t* p = base + r * rstride + 8 * g;
  FragU u;
  u.u[0] = *(const u32x4*)(p);
  u.u[1] = *(const u32x4*)(p + 16);
  return u.f;
}

#ifdef USE_TDM
// -------------------------------------------------------------------------
// TDM 2-D tile load: global (row-major, 2-byte elems, row stride rs elems)
//   -> LDS at byte offset lds_addr, tile tile_d0 x tile_d1 (elems x rows),
// LDS rows padded by 16 bytes after each 32-elem (64-byte) row so the LDS
// row stride stays 40 elements.  tens_d0/tens_d1 = remaining tensor extent
// from the tile start; rows/cols beyond it are zero-filled by the TDM.
// D# layout per CDNA5 ISA 8.3/8.4.  Issue from ONE wave; EXEC is ignored.
// This toolchain exposes the 6-arg builtin:
//   (u32x4 g0, i32x8 g1, i32x4 g2, i32x4 g3, i32x8 gx, i32 cpol)
// -------------------------------------------------------------------------
__device__ __forceinline__ void tdm_load_2d(unsigned lds_addr,
                                            const void* gptr,
                                            unsigned tile_d0, unsigned tile_d1,
                                            unsigned tens_d0, unsigned tens_d1,
                                            unsigned rs) {
  unsigned long long ga = (unsigned long long)gptr;
  u32x4 g0;
  g0[0] = 1u;                                   // count=1, user descriptor
  g0[1] = lds_addr;                             // LDS byte address
  g0[2] = (unsigned)ga;                         // global_addr[31:0]
  g0[3] = ((unsigned)(ga >> 32) & 0x01FFFFFFu)  // global_addr[56:32]
          | (2u << 30);                         // type = 2 ("image")
  i32x8 g1;
  // data_size=1 (2B), pad_enable=1, pad_interval=3 (16 DW), pad_amount=3 (4 DW)
  g1[0] = (int)((1u << 16) | (1u << 20) | (3u << 22) | (3u << 25));
  g1[1] = (int)((tens_d0 & 0xFFFFu) << 16);             // tensor_dim0[15:0]
  g1[2] = (int)(((tens_d0 >> 16) & 0xFFFFu) |
                ((tens_d1 & 0xFFFFu) << 16));           // d0[31:16] | d1[15:0]
  g1[3] = (int)(((tens_d1 >> 16) & 0xFFFFu) |
                ((tile_d0 & 0xFFFFu) << 16));           // d1[31:16] | tile_d0
  g1[4] = (int)(tile_d1 & 0xFFFFu);                     // tile_dim1, tile_dim2=0
  g1[5] = (int)rs;                                      // tensor_dim0_stride
  g1[6] = 0;
  g1[7] = 0;
  i32x4 gz4 = {0, 0, 0, 0};
  i32x8 gz8 = {0, 0, 0, 0, 0, 0, 0, 0};
  __builtin_amdgcn_tensor_load_to_lds(g0, g1, gz4, gz4, gz8, 0);
}
#endif

// ---------------- fp32 -> bf16 convert ----------------
__global__ __launch_bounds__(256) void cvt_kernel(const float* __restrict__ s,
                                                  bf16_t* __restrict__ d,
                                                  size_t n) {
  size_t i  = (size_t)blockIdx.x * blockDim.x + threadIdx.x;
  size_t st = (size_t)gridDim.x * blockDim.x;
  for (; i < n; i += st) d[i] = (bf16_t)s[i];
}

// ---------------- LayerNorm (optionally gathers embedding rows) ----------
__global__ __launch_bounds__(256) void ln_kernel(
    const float* __restrict__ src, const float* __restrict__ emb,
    const int* __restrict__ ids, const float* __restrict__ g,
    const float* __restrict__ b, float* __restrict__ of,
    bf16_t* __restrict__ ob) {
  __shared__ float r1[256], r2[256];
  const int row = blockIdx.x, tid = threadIdx.x;
  const float* x = ids ? (emb + (size_t)ids[row] * EE)
                       : (src + (size_t)row * EE);
  f32x4 v = ((const f32x4*)x)[tid];
  float s  = v[0] + v[1] + v[2] + v[3];
  float ss = v[0]*v[0] + v[1]*v[1] + v[2]*v[2] + v[3]*v[3];
  r1[tid] = s; r2[tid] = ss; __syncthreads();
  for (int o = 128; o > 0; o >>= 1) {
    if (tid < o) { r1[tid] += r1[tid + o]; r2[tid] += r2[tid + o]; }
    __syncthreads();
  }
  const float mean = r1[0] * (1.f / EE);
  const float var  = r2[0] * (1.f / EE) - mean * mean;
  const float rstd = rsqrtf(var + 1e-5f);
  const f32x4 gv = ((const f32x4*)g)[tid];
  const f32x4 bv = ((const f32x4*)b)[tid];
  f32x4 y;
  bf16x4 yb;
#pragma unroll
  for (int j = 0; j < 4; ++j) {
    y[j]  = (v[j] - mean) * rstd * gv[j] + bv[j];
    yb[j] = (bf16_t)y[j];
  }
  if (of) ((f32x4*)(of + (size_t)row * EE))[tid] = y;
  if (ob) ((bf16x4*)(ob + (size_t)row * EE))[tid] = yb;
}

// ---------------- generic WMMA GEMM:  C[M,N] = A[M,K] * W[N,K]^T ----------
// 256 threads = 8 waves; macro tile 64(M) x 128(N); K step 32.
// Tiles staged into LDS by the Tensor Data Mover (wave 0 issues, waits on
// TENSORcnt, workgroup barrier publishes).  Fallback: vectorized staging.
template <bool BIAS, bool RES, bool GELU, bool WF32, bool WBF16>
__global__ __launch_bounds__(256) void gemm_bf16_kernel(
    const bf16_t* __restrict__ A, const bf16_t* __restrict__ W,
    const float* __restrict__ bias, const float* __restrict__ Res,
    float* __restrict__ Cf, bf16_t* __restrict__ Cb,
    int M, int N, int K) {
  __shared__ bf16_t As[64 * 40];
  __shared__ bf16_t Ws[128 * 40];
  const int tid = threadIdx.x;
  const int lane = tid & 31, wave = tid >> 5;
  const int wr = wave & 3, wc = wave >> 2;
  const int bm = blockIdx.y * 64, bn = blockIdx.x * 128;

  v8f acc[4] = {};

#ifdef USE_TDM
  const unsigned as_lds = (unsigned)(size_t)(void*)&As[0];
  const unsigned ws_lds = (unsigned)(size_t)(void*)&Ws[0];
  const unsigned remM = (unsigned)(M - bm);
  const unsigned remN = (unsigned)((N - bn) > 0 ? (N - bn) : 0);
#endif

  for (int k0 = 0; k0 < K; k0 += 32) {
#ifdef USE_TDM
    if (wave == 0) {
      tdm_load_2d(as_lds, A + (size_t)bm * K + k0, 32, 64,
                  (unsigned)(K - k0), remM, (unsigned)K);
      tdm_load_2d(ws_lds, W + (size_t)bn * K + k0, 32, 128,
                  (unsigned)(K - k0), remN, (unsigned)K);
      __builtin_amdgcn_s_wait_tensorcnt(0);
    }
#else
    const u32x4 zero4 = {0u, 0u, 0u, 0u};
    {  // stage A tile: 64 rows x 32, one 16B chunk per thread
      int row = tid >> 2, kk = (tid & 3) * 8;
      int gr = bm + row;
      const u32x4* src = (const u32x4*)(A + (size_t)gr * K + k0 + kk);
      *(u32x4*)&As[row * 40 + kk] = (gr < M) ? *src : zero4;
    }
    {  // stage W tile: 128 rows x 32, two 16B chunks per thread
      int row = tid >> 1, kk = (tid & 1) * 16;
      int gn = bn + row;
      const u32x4* src = (const u32x4*)(W + (size_t)gn * K + k0 + kk);
      u32x4* dst = (u32x4*)&Ws[row * 40 + kk];
      dst[0] = (gn < N) ? src[0] : zero4;
      dst[1] = (gn < N) ? src[1] : zero4;
    }
#endif
    __syncthreads();
    v16bf a = load_frag_v(&As[(wr * 16) * 40], 40);
#pragma unroll
    for (int i = 0; i < 4; ++i) {
      v16bf bfr = load_frag_v(&Ws[(wc * 64 + i * 16) * 40], 40);
      acc[i] = __builtin_amdgcn_wmma_f32_16x16x32_bf16(
          false, a, false, bfr, (short)0, acc[i], false, false);
    }
    __syncthreads();
  }

  // epilogue (C layout: lane = n + 16*(m>=8), vgpr r -> m = r + 8*g)
  const int g = lane >> 4, n0 = lane & 15;
#pragma unroll
  for (int i = 0; i < 4; ++i) {
    int col = bn + wc * 64 + i * 16 + n0;
    if (col < N) {
#pragma unroll
      for (int r = 0; r < 8; ++r) {
        int row = bm + wr * 16 + r + 8 * g;
        if (row < M) {
          float v = acc[i][r];
          if (BIAS) v += bias[col];
          if (GELU) v = 0.5f * v * (1.f + erff(v * 0.70710678118654752f));
          if (RES)  v += Res[(size_t)row * N + col];
          if (WF32) Cf[(size_t)row * N + col] = v;
          if (WBF16) Cb[(size_t)row * N + col] = (bf16_t)v;
        }
      }
    }
  }
}

// ---------------- RoPE: qkv f32 [BT,3,H,HD] ->
//   q,k bf16 [B,H,T,HD]  and  v bf16 TRANSPOSED [B,H,HD,T] ----------------
__global__ __launch_bounds__(256) void rope_kernel(
    const float* __restrict__ qkv, bf16_t* __restrict__ qb,
    bf16_t* __restrict__ kb, bf16_t* __restrict__ vtb) {
  const int row = blockIdx.x;           // b*T + t
  const int b = row >> 9, t = row & 511;
  const float* src = qkv + (size_t)row * (3 * EE);
  for (int idx = threadIdx.x; idx < 3 * EE; idx += 256) {
    int s = idx >> 10, rem = idx & 1023;
    int h = rem >> 6, d = rem & 63;
    int bh = b * HH + h;
    float x = src[idx];
    if (s == 2) {
      vtb[((size_t)bh * HD + d) * TT + t] = (bf16_t)x;   // vT[b,h,d,t]
    } else {
      int dm = d & 31;
      float ang = (float)t * __powf(10000.f, -(float)(2 * dm) * (1.f / HD));
      float c = __cosf(ang), sn = __sinf(ang);
      float partner = src[s * EE + h * HD + ((d < 32) ? d + 32 : d - 32)];
      float rot = (d < 32) ? -partner : partner;
      float val = x * c + rot * sn;
      (s == 0 ? qb : kb)[((size_t)bh * TT + t) * HD + d] = (bf16_t)val;
    }
  }
}

// ---------------- fused flash attention: one wave per (b,h, 16-row qblock) -
__global__ __launch_bounds__(32) void attn_kernel(
    const bf16_t* __restrict__ qb, const bf16_t* __restrict__ kb,
    const bf16_t* __restrict__ vtb, const unsigned char* __restrict__ mask,
    bf16_t* __restrict__ obf) {
  __shared__ float  Sld[16][33];
  __shared__ bf16_t Pld[16][40];
  __shared__ float  rowM[16], rowL[16], rowA[16];

  const int blk = blockIdx.x;
  const int qb0 = blk & 31;          // q block (T/16 = 32)
  const int bh  = blk >> 5;          // b*H + h
  const int h   = bh & 15, b = bh >> 4;
  const int lane = threadIdx.x, g = lane >> 4, n0 = lane & 15;

  const bf16_t* Q  = qb  + ((size_t)bh * TT + qb0 * 16) * HD;
  const bf16_t* K  = kb  + (size_t)bh * TT * HD;
  const bf16_t* VT = vtb + (size_t)bh * HD * TT;   // [HD, T]

  v16bf aq0 = load_frag_v(Q, HD);        // K-dim 0..31
  v16bf aq1 = load_frag_v(Q + 32, HD);   // K-dim 32..63
  v8f o[4] = {};
  if (lane < 16) { rowM[lane] = -1e30f; rowL[lane] = 0.f; }
  __syncthreads();

  for (int j = 0; j < 16; ++j) {          // 32-key chunks
    const bf16_t* Kj = K + (size_t)j * 32 * HD;
#pragma unroll
    for (int i = 0; i < 2; ++i) {         // S tile: 16 x 32 keys
      v8f s = {};
      v16bf b0 = load_frag_v(Kj + i * 16 * HD, HD);
      v16bf b1 = load_frag_v(Kj + i * 16 * HD + 32, HD);
      s = __builtin_amdgcn_wmma_f32_16x16x32_bf16(false, aq0, false, b0,
                                                  (short)0, s, false, false);
      s = __builtin_amdgcn_wmma_f32_16x16x32_bf16(false, aq1, false, b1,
                                                  (short)0, s, false, false);
#pragma unroll
      for (int r = 0; r < 8; ++r) Sld[r + 8 * g][i * 16 + n0] = s[r];
    }
    __syncthreads();
    if (lane < 16) {                      // online softmax bookkeeping
      const int t = lane;
      float mx = rowM[t];
      float sv[32];
#pragma unroll
      for (int c = 0; c < 32; ++c) {
        float madd = mask[b * TT + j * 32 + c] ? 0.f : -10000.f;
        float v = Sld[t][c] * 0.125f + madd;   // scale = HD^-0.5
        sv[c] = v;
        mx = fmaxf(mx, v);
      }
      float alpha = __expf(rowM[t] - mx);
      float l = rowL[t] * alpha;
#pragma unroll
      for (int c = 0; c < 32; ++c) {
        float e = __expf(sv[c] - mx);
        l += e;
        Pld[t][c] = (bf16_t)e;
      }
      rowM[t] = mx; rowL[t] = l; rowA[t] = alpha;
    }
    __syncthreads();
    v16bf ap = load_frag_v(&Pld[0][0], 40);
#pragma unroll
    for (int i = 0; i < 4; ++i) {         // O += P(16x32) * V(32x64)
#pragma unroll
      for (int r = 0; r < 8; ++r) o[i][r] *= rowA[r + 8 * g];
      v16bf bv = load_frag_v(VT + (size_t)(i * 16) * TT + j * 32, TT);
      o[i] = __builtin_amdgcn_wmma_f32_16x16x32_bf16(false, ap, false, bv,
                                                     (short)0, o[i], false,
                                                     false);
    }
    __syncthreads();
  }
#pragma unroll
  for (int i = 0; i < 4; ++i)
#pragma unroll
    for (int r = 0; r < 8; ++r) {
      int m = r + 8 * g;
      float val = o[i][r] / rowL[m];
      int t = qb0 * 16 + m;
      obf[((size_t)b * TT + t) * EE + h * HD + i * 16 + n0] = (bf16_t)val;
    }
}

// ---------------- pooling: qp = Wq * pool_q + bq ----------------
__global__ __launch_bounds__(256) void qp_kernel(
    const float* __restrict__ pool_q, const float* __restrict__ pool_in_w,
    const float* __restrict__ pool_in_b, float* __restrict__ qp) {
  int e = blockIdx.x * 256 + threadIdx.x;
  if (e < EE) {
    float s = pool_in_b[e];
    const float* w = pool_in_w + (size_t)e * EE;
    for (int k = 0; k < EE; ++k) s += pool_q[k] * w[k];
    qp[e] = s;
  }
}

// ---------------- pooling: sc[b,t] = dot(qp, kp[b,t]) * E^-0.5 (+mask) ---
__global__ __launch_bounds__(256) void score_kernel(
    const float* __restrict__ qp, const float* __restrict__ kp,
    const unsigned char* __restrict__ mask, float* __restrict__ sc) {
  __shared__ float red[256];
  const int row = blockIdx.x, tid = threadIdx.x;
  const f32x4* k = (const f32x4*)(kp + (size_t)row * EE);
  const f32x4* q4 = (const f32x4*)qp;
  f32x4 a = q4[tid], bq = k[tid];
  red[tid] = a[0]*bq[0] + a[1]*bq[1] + a[2]*bq[2] + a[3]*bq[3];
  __syncthreads();
  for (int o = 128; o > 0; o >>= 1) {
    if (tid < o) red[tid] += red[tid + o];
    __syncthreads();
  }
  if (tid == 0) sc[row] = mask[row] ? red[0] * 0.03125f : -INFINITY;
}

// ---------------- pooling softmax over T=512 per batch ----------------
__global__ __launch_bounds__(256) void softmax512_kernel(float* __restrict__ sc) {
  __shared__ float red[256];
  const int tid = threadIdx.x;
  float* r = sc + (size_t)blockIdx.x * TT;
  float a0 = r[tid], a1 = r[tid + 256];
  red[tid] = fmaxf(a0, a1); __syncthreads();
  for (int o = 128; o > 0; o >>= 1) {
    if (tid < o) red[tid] = fmaxf(red[tid], red[tid + o]);
    __syncthreads();
  }
  float mx = red[0]; __syncthreads();
  float e0 = __expf(a0 - mx), e1 = __expf(a1 - mx);
  red[tid] = e0 + e1; __syncthreads();
  for (int o = 128; o > 0; o >>= 1) {
    if (tid < o) red[tid] += red[tid + o];
    __syncthreads();
  }
  float inv = 1.f / red[0];
  r[tid] = e0 * inv; r[tid + 256] = e1 * inv;
}

// ---------------- pooled[b,e] = sum_t aw[b,t] * vp[b,t,e] ----------------
__global__ __launch_bounds__(256) void pool_sum_kernel(
    const float* __restrict__ aw, const float* __restrict__ vp,
    bf16_t* __restrict__ pooled) {
  const int b = blockIdx.x;
  const int e = blockIdx.y * 256 + threadIdx.x;
  float s = 0.f;
  for (int t = 0; t < TT; ++t)
    s += aw[b * TT + t] * vp[((size_t)b * TT + t) * EE + e];
  pooled[(size_t)b * EE + e] = (bf16_t)s;
}

// =========================================================================
extern "C" void kernel_launch(void* const* d_in, const int* in_sizes, int n_in,
                              void* d_out, int out_size, void* d_ws,
                              size_t ws_size, hipStream_t stream) {
  (void)in_sizes; (void)n_in; (void)out_size;
  // ---- inputs (setup_inputs order) ----
  const int*   input_ids = (const int*)d_in[0];
  const unsigned char* amask = (const unsigned char*)d_in[1];
  const float* emb       = (const float*)d_in[2];
  const float* in_g      = (const float*)d_in[3];
  const float* in_b      = (const float*)d_in[4];
  const float* ln1_g     = (const float*)d_in[5];
  const float* ln1_b     = (const float*)d_in[6];
  const float* qkv_w     = (const float*)d_in[7];
  const float* ao_w      = (const float*)d_in[8];
  const float* ao_b      = (const float*)d_in[9];
  const float* ln2_g     = (const float*)d_in[10];
  const float* ln2_b     = (const float*)d_in[11];
  const float* w1        = (const float*)d_in[12];
  const float* b1        = (const float*)d_in[13];
  const float* w2        = (const float*)d_in[14];
  const float* b2        = (const float*)d_in[15];
  const float* fin_g     = (const float*)d_in[16];
  const float* fin_b     = (const float*)d_in[17];
  const float* pool_q    = (const float*)d_in[18];
  const float* pin_w     = (const float*)d_in[19];
  const float* pin_b     = (const float*)d_in[20];
  const float* pout_w    = (const float*)d_in[21];
  const float* pout_b    = (const float*)d_in[22];
  const float* pln_g     = (const float*)d_in[23];
  const float* pln_b     = (const float*)d_in[24];
  const float* proj_w    = (const float*)d_in[25];
  const float* proj_b    = (const float*)d_in[26];
  float* out = (float*)d_out;

  // ---- workspace bump allocator ----
  char* base = (char*)d_ws;
  size_t off = 0;
  auto alloc = [&](size_t bytes) -> void* {
    off = (off + 255) & ~(size_t)255;
    void* p = base + off;
    off += bytes;
    return p;
  };
  const size_t NQKV = (size_t)LL * 3 * EE * EE;
  const size_t NAO  = (size_t)LL * EE * EE;
  const size_t NW1  = (size_t)LL * FF * EE;
  const size_t NW2  = (size_t)LL * EE * FF;
  bf16_t* wqkv_bf = (bf16_t*)alloc(NQKV * 2);
  bf16_t* wo_bf   = (bf16_t*)alloc(NAO * 2);
  bf16_t* w1_bf   = (bf16_t*)alloc(NW1 * 2);
  bf16_t* w2_bf   = (bf16_t*)alloc(NW2 * 2);
  bf16_t* wpin_bf = (bf16_t*)alloc((size_t)3 * EE * EE * 2);
  bf16_t* wpout_bf= (bf16_t*)alloc((size_t)EE * EE * 2);
  bf16_t* wproj_bf= (bf16_t*)alloc((size_t)OO * EE * 2);
  float*  xa      = (float*) alloc((size_t)BT * EE * 4);
  float*  xb      = (float*) alloc((size_t)BT * EE * 4);
  bf16_t* xn_bf   = (bf16_t*)alloc((size_t)BT * EE * 2);
  float*  qkvf    = (float*) alloc((size_t)BT * 3 * EE * 4);
  bf16_t* qbuf    = (bf16_t*)alloc((size_t)BT * EE * 2);
  bf16_t* kbuf    = (bf16_t*)alloc((size_t)BT * EE * 2);
  bf16_t* vtbuf   = (bf16_t*)alloc((size_t)BT * EE * 2);
  bf16_t* obf     = (bf16_t*)alloc((size_t)BT * EE * 2);
  bf16_t* hbf     = (bf16_t*)alloc((size_t)BT * FF * 2);
  float*  kp      = (float*) alloc((size_t)BT * EE * 4);
  float*  vp      = (float*) alloc((size_t)BT * EE * 4);
  float*  qp      = (float*) alloc((size_t)EE * 4);
  float*  sc      = (float*) alloc((size_t)BB * TT * 4);
  bf16_t* pooled_bf   = (bf16_t*)alloc((size_t)BB * EE * 2);
  float*  pooled2     = (float*) alloc((size_t)BB * EE * 4);
  bf16_t* pooledln_bf = (bf16_t*)alloc((size_t)BB * EE * 2);
  if (off > ws_size) return;  // workspace too small; nothing sane to do

  // ---- convert all GEMM weights to bf16 ----
  cvt_kernel<<<4096, 256, 0, stream>>>(qkv_w, wqkv_bf, NQKV);
  cvt_kernel<<<4096, 256, 0, stream>>>(ao_w, wo_bf, NAO);
  cvt_kernel<<<4096, 256, 0, stream>>>(w1, w1_bf, NW1);
  cvt_kernel<<<4096, 256, 0, stream>>>(w2, w2_bf, NW2);
  cvt_kernel<<<2048, 256, 0, stream>>>(pin_w, wpin_bf, (size_t)3 * EE * EE);
  cvt_kernel<<<1024, 256, 0, stream>>>(pout_w, wpout_bf, (size_t)EE * EE);
  cvt_kernel<<<1024, 256, 0, stream>>>(proj_w, wproj_bf, (size_t)OO * EE);

  const dim3 blk256(256);
  // embedding gather + input LN -> residual stream xa (f32)
  ln_kernel<<<BT, blk256, 0, stream>>>(nullptr, emb, input_ids, in_g, in_b,
                                       xa, nullptr);

  float* xcur = xa;   // residual stream
  float* xnext = xb;
  for (int l = 0; l < LL; ++l) {
    // LN1 -> xn_bf
    ln_kernel<<<BT, blk256, 0, stream>>>(xcur, nullptr, nullptr,
                                         ln1_g + l * EE, ln1_b + l * EE,
                                         nullptr, xn_bf);
    // qkv = xn @ Wqkv^T      [8192,3072]
    gemm_bf16_kernel<false, false, false, true, false>
        <<<dim3((3 * EE + 127) / 128, (BT + 63) / 64), blk256, 0, stream>>>(
            xn_bf, wqkv_bf + (size_t)l * 3 * EE * EE, nullptr, nullptr,
            qkvf, nullptr, BT, 3 * EE, EE);
    // RoPE + layout: q,k -> [B,H,T,HD]; v -> [B,H,HD,T]
    rope_kernel<<<BT, blk256, 0, stream>>>(qkvf, qbuf, kbuf, vtbuf);
    // fused attention -> obf [8192,1024] bf16
    attn_kernel<<<BB * HH * (TT / 16), dim3(32), 0, stream>>>(
        qbuf, kbuf, vtbuf, amask, obf);
    // x = x + o @ Wo^T + bo  -> xnext (f32)
    gemm_bf16_kernel<true, true, false, true, false>
        <<<dim3((EE + 127) / 128, (BT + 63) / 64), blk256, 0, stream>>>(
            obf, wo_bf + (size_t)l * EE * EE, ao_b + l * EE, xcur,
            xnext, nullptr, BT, EE, EE);
    // LN2 -> xn_bf
    ln_kernel<<<BT, blk256, 0, stream>>>(xnext, nullptr, nullptr,
                                         ln2_g + l * EE, ln2_b + l * EE,
                                         nullptr, xn_bf);
    // h = gelu(xn @ W1^T + b1) -> hbf (bf16)
    gemm_bf16_kernel<true, false, true, false, true>
        <<<dim3((FF + 127) / 128, (BT + 63) / 64), blk256, 0, stream>>>(
            xn_bf, w1_bf + (size_t)l * FF * EE, b1 + l * FF, nullptr,
            nullptr, hbf, BT, FF, EE);
    // x = x + h @ W2^T + b2 -> xcur (f32)
    gemm_bf16_kernel<true, true, false, true, false>
        <<<dim3((EE + 127) / 128, (BT + 63) / 64), blk256, 0, stream>>>(
            hbf, w2_bf + (size_t)l * EE * FF, b2 + l * EE, xnext,
            xcur, nullptr, BT, EE, FF);
  }

  // final LN -> xn_bf
  ln_kernel<<<BT, blk256, 0, stream>>>(xcur, nullptr, nullptr, fin_g, fin_b,
                                       nullptr, xn_bf);
  // pooling: qp, kp, vp
  qp_kernel<<<4, blk256, 0, stream>>>(pool_q, pin_w, pin_b, qp);
  gemm_bf16_kernel<true, false, false, true, false>
      <<<dim3(8, (BT + 63) / 64), blk256, 0, stream>>>(
          xn_bf, wpin_bf + (size_t)EE * EE, pin_b + EE, nullptr,
          kp, nullptr, BT, EE, EE);
  gemm_bf16_kernel<true, false, false, true, false>
      <<<dim3(8, (BT + 63) / 64), blk256, 0, stream>>>(
          xn_bf, wpin_bf + (size_t)2 * EE * EE, pin_b + 2 * EE, nullptr,
          vp, nullptr, BT, EE, EE);
  score_kernel<<<BT, blk256, 0, stream>>>(qp, kp, amask, sc);
  softmax512_kernel<<<BB, blk256, 0, stream>>>(sc);
  pool_sum_kernel<<<dim3(BB, EE / 256), blk256, 0, stream>>>(sc, vp, pooled_bf);
  // pooled @ pool_out_w^T + b  -> pooled2 (f32)
  gemm_bf16_kernel<true, false, false, true, false>
      <<<dim3(8, 1), blk256, 0, stream>>>(
          pooled_bf, wpout_bf, pout_b, nullptr, pooled2, nullptr, BB, EE, EE);
  // pool LN -> pooledln_bf
  ln_kernel<<<BB, blk256, 0, stream>>>(pooled2, nullptr, nullptr, pln_g,
                                       pln_b, nullptr, pooledln_bf);
  // final projection -> d_out (f32)
  gemm_bf16_kernel<true, false, false, true, false>
      <<<dim3(8, 1), blk256, 0, stream>>>(
          pooledln_bf, wproj_bf, proj_b, nullptr, out, nullptr, BB, OO, EE);
}
